// Decoder_23210003268185
// MI455X (gfx1250) — matrix operand order repaired and checked by reading, not compiled, not run on hardware
//
#include <hip/hip_runtime.h>
#include <hip/hip_bf16.h>
#include <stdint.h>

// ---------------- CDNA5 WMMA types ----------------
typedef __attribute__((ext_vector_type(16))) _Float16 v16h;
typedef __attribute__((ext_vector_type(8)))  float    v8f;

// ---------------- problem dims ----------------
constexpr int CB  = 32;    // batch
constexpr int CTE = 1024;  // encoder length
constexpr int CTD = 128;   // decoder length
constexpr int CE  = 512;   // encoder dim
constexpr int CD  = 1024;  // LSTM hidden
constexpr int CA  = 128;   // attention dim
constexpr int CP  = 256;   // prenet dim
constexpr int CM  = 80;    // n_mels
constexpr int CLF = 32;    // location filters
constexpr int CLK = 31;    // location kernel
constexpr int CG  = 4096;  // 4*CD gates
constexpr int CKA = CP + CE; // 768 attn-LSTM input dim

__device__ __forceinline__ float sigmoidf_(float x) { return 1.0f / (1.0f + expf(-x)); }
__device__ __forceinline__ unsigned wang_(unsigned x) {
  x ^= x >> 16; x *= 0x7feb352dU; x ^= x >> 15; x *= 0x846ca68bU; x ^= x >> 16; return x;
}

// ---- CDNA5 async global -> LDS copy (GLOBAL_LOAD_ASYNC_TO_LDS_B128, ASYNCcnt) ----
__device__ __forceinline__ void async_g2l_b128(uint32_t lds_off, const void* g) {
  unsigned long long ga = (unsigned long long)g;
  asm volatile("global_load_async_to_lds_b128 %0, %1, off"
               :: "v"(lds_off), "v"(ga) : "memory");
}
__device__ __forceinline__ void wait_asynccnt0() {
  asm volatile("s_wait_asynccnt 0" ::: "memory");
}

// ---- A-fragment (16x32 f16, row-major source, ld = ldk halves) ----
// lanes 0-15: M=lane, K 0-7 (v0-3) and 16-23 (v4-7); lanes 16-31: M=lane-16, K 8-15 / 24-31
__device__ __forceinline__ v16h load_a_f16(const _Float16* X, int ldk, int mbase, int ktile, int lane) {
  int m  = mbase + (lane & 15);
  int k0 = ktile * 32 + ((lane & 16) ? 8 : 0);
  const _Float16* p = X + (size_t)m * ldk + k0;
  union { v16h v; uint4 q[2]; } u;
  u.q[0] = *(const uint4*)(p);        // K k0..k0+7
  u.q[1] = *(const uint4*)(p + 16);   // K k0+16..k0+23
  return u.v;
}
// same fragment but converting from an f32 row-major source on the fly
__device__ __forceinline__ v16h load_a_f32(const float* X, int ldk, int mbase, int ktile, int lane) {
  int m  = mbase + (lane & 15);
  int k0 = ktile * 32 + ((lane & 16) ? 8 : 0);
  const float* p = X + (size_t)m * ldk + k0;
  union { v16h v; _Float16 h[16]; } u;
#pragma unroll
  for (int j = 0; j < 8; ++j) { u.h[j] = (_Float16)p[j]; u.h[8 + j] = (_Float16)p[16 + j]; }
  return u.v;
}
// ---- B-fragment from pre-swizzled tiles: tile = 512 halves, lane-major (16 halves/lane) ----
__device__ __forceinline__ v16h load_b_tile(const _Float16* Wt, int tile, int lane) {
  union { v16h v; uint4 q[2]; } u;
  const uint4* p = (const uint4*)(Wt + (size_t)tile * 512 + (size_t)lane * 16);
  u.q[0] = p[0]; u.q[1] = p[1];
  return u.v;
}

// ================= prep kernels =================
// Swizzle W[N,K] f32 -> f16 B-fragment tiles (tile = ntile*(K/32)+ktile).
__global__ void k_swizzle(const float* __restrict__ W, _Float16* __restrict__ out, int N, int K) {
  int tid = blockIdx.x * blockDim.x + threadIdx.x;
  int KT = K / 32;
  int total = (N / 16) * KT * 32;
  if (tid >= total) return;
  int lane = tid & 31, tile = tid >> 5;
  int ktile = tile % KT, ntile = tile / KT;
  int n  = ntile * 16 + (lane & 15);
  int kb = ktile * 32 + ((lane & 16) ? 8 : 0);
  _Float16* o = out + (size_t)tile * 512 + (size_t)lane * 16;
  const float* w = W + (size_t)n * K + kb;
#pragma unroll
  for (int j = 0; j < 8; ++j) { o[j] = (_Float16)w[j]; o[8 + j] = (_Float16)w[16 + j]; }
}

// Fold loc conv (LF,1,LK) with loc_w (A,LF): U[a][k] = sum_f loc_w[a,f]*conv[f,0,k]
__global__ void k_uloc(const float* __restrict__ loc_w, const float* __restrict__ convw,
                       float* __restrict__ U) {
  int a = threadIdx.x;  // 128
  for (int k = 0; k < CLK; ++k) {
    float s = 0.f;
    for (int f = 0; f < CLF; ++f) s += loc_w[a * CLF + f] * convw[f * CLK + k];
    U[a * CLK + k] = s;
  }
}

// processed_memory = memory @ mem_w.T : [CB*CTE, CE] x [CE, CA] (WMMA, one-time)
// Dual M-tile accumulators per wave: independent WMMAs interleave -> no RAW NOP bubbles.
__global__ void k_pm_wmma(const float* __restrict__ memory, const _Float16* __restrict__ wmem_t,
                          float* __restrict__ pm) {
  int lane  = threadIdx.x & 31;
  int ntile = threadIdx.x >> 5;      // 0..7 (CA/16)
  int mbase = blockIdx.x * 32;       // 1024 blocks, 32 rows each
  v8f c0 = {}, c1 = {};
  const int KT = CE / 32;            // 16
  for (int kt = 0; kt < KT; ++kt) {
    v16h b  = load_b_tile(wmem_t, ntile * KT + kt, lane);
    v16h a0 = load_a_f32(memory, CE, mbase, kt, lane);
    v16h a1 = load_a_f32(memory, CE, mbase + 16, kt, lane);
    c0 = __builtin_amdgcn_wmma_f32_16x16x32_f16(false, a0, false, b, (short)0, c0, false, false);
    c1 = __builtin_amdgcn_wmma_f32_16x16x32_f16(false, a1, false, b, (short)0, c1, false, false);
  }
  int n = ntile * 16 + (lane & 15);
  int mlo = (lane & 16) ? 8 : 0;
#pragma unroll
  for (int r = 0; r < 8; ++r) {
    pm[(size_t)(mbase + mlo + r) * CA + n]      = c0[r];
    pm[(size_t)(mbase + 16 + mlo + r) * CA + n] = c1[r];
  }
}

__global__ void k_init(float* ah, float* ac, float* dh, float* dc,
                       _Float16* ah16, _Float16* dh16, float* cumw) {
  int i = blockIdx.x * 256 + threadIdx.x;  // covers CB*CD == CB*CTE == 32768
  if (i < CB * CD) {
    ah[i] = 0.f; ac[i] = 0.f; dh[i] = 0.f; dc[i] = 0.f;
    ah16[i] = (_Float16)0.f; dh16[i] = (_Float16)0.f;
  }
  if (i < CB * CTE) cumw[i] = ((i % CTE) == 0) ? 0.1f : 0.f;
}

// ================= per-step kernels =================
__global__ void k_prenet(const float* __restrict__ dec_inputs, const float* __restrict__ w1,
                         const float* __restrict__ w2, _Float16* __restrict__ xattn, int t) {
  __shared__ float x1[CP];
  int b = blockIdx.x, p = threadIdx.x;  // 256 threads
  float s = 0.f;
  if (t > 0) {
    const float* xin = dec_inputs + (size_t)b * CM * CTD + (t - 1);
    const float* wr = w1 + (size_t)p * CM;
    for (int m = 0; m < CM; ++m) s += wr[m] * xin[(size_t)m * CTD];
  }
  s = fmaxf(s, 0.f);
  s = (wang_(((2u * t) * 7919u + b) * 257u + p) & 1u) ? s * 2.f : 0.f;  // dropout p=0.5
  x1[p] = s;
  __syncthreads();
  float s2 = 0.f;
  const float* w2r = w2 + (size_t)p * CP;
  for (int k = 0; k < CP; ++k) s2 += w2r[k] * x1[k];
  s2 = fmaxf(s2, 0.f);
  s2 = (wang_(((2u * t + 1u) * 7919u + b) * 257u + p) & 1u) ? s2 * 2.f : 0.f;
  xattn[(size_t)b * CKA + p] = (_Float16)s2;
}

// Location-sensitive attention: energies + masked softmax + cumulative weights (one block per b)
__global__ void k_attention(const float* __restrict__ ah, const float* __restrict__ q_w,
                            const float* __restrict__ pm, const float* __restrict__ U,
                            const float* __restrict__ v_w, const float* __restrict__ v_b,
                            const unsigned char* __restrict__ mask,
                            float* __restrict__ cumw, float* __restrict__ aligns, int t) {
  __shared__ float q[CA];
  __shared__ float e[CTE];
  __shared__ float red[256];
  int b = blockIdx.x, tid = threadIdx.x;
  if (tid < CA) {
    float s = 0.f;
    const float* hr = ah + (size_t)b * CD;
    const float* wr = q_w + (size_t)tid * CD;
    for (int d = 0; d < CD; ++d) s += hr[d] * wr[d];
    q[tid] = s;
  }
  __syncthreads();
  float vb = v_b[0];
  const float* cw = cumw + (size_t)b * CTE;
  for (int i = 0; i < 4; ++i) {
    int tc = tid + i * 256;
    float win[CLK];
#pragma unroll
    for (int k = 0; k < CLK; ++k) {
      int src = tc + k - (CLK - 1) / 2;
      win[k] = (src >= 0 && src < CTE) ? cw[src] : 0.f;
    }
    const float* pmr = pm + ((size_t)b * CTE + tc) * CA;
    float en = vb;
    for (int a = 0; a < CA; ++a) {
      float s = q[a] + pmr[a];
      const float* Ur = U + a * CLK;
#pragma unroll
      for (int k = 0; k < CLK; ++k) s += Ur[k] * win[k];
      en += v_w[a] * tanhf(s);
    }
    if (mask[(size_t)b * CTE + tc]) en = -INFINITY;
    e[tc] = en;
  }
  __syncthreads();
  float mx = -INFINITY;
  for (int i = 0; i < 4; ++i) mx = fmaxf(mx, e[tid + i * 256]);
  red[tid] = mx; __syncthreads();
  for (int s = 128; s > 0; s >>= 1) { if (tid < s) red[tid] = fmaxf(red[tid], red[tid + s]); __syncthreads(); }
  mx = red[0]; __syncthreads();
  float sum = 0.f;
  for (int i = 0; i < 4; ++i) { int tc = tid + i * 256; float ex = expf(e[tc] - mx); e[tc] = ex; sum += ex; }
  red[tid] = sum; __syncthreads();
  for (int s = 128; s > 0; s >>= 1) { if (tid < s) red[tid] += red[tid + s]; __syncthreads(); }
  float inv = 1.f / red[0];
  float* awout = aligns + ((size_t)b * CTD + t) * CTE;
  float* cwm = cumw + (size_t)b * CTE;
  for (int i = 0; i < 4; ++i) {
    int tc = tid + i * 256;
    float aw = e[tc] * inv;
    awout[tc] = aw;
    cwm[tc] += aw;
  }
}

// ctx[b,e] = sum_tc aw[b,tc] * memory[b,tc,e]   (32 independent GEMVs; L2-resident memory)
__global__ void k_context(const float* __restrict__ aligns, const float* __restrict__ memory,
                          float* __restrict__ ctx, _Float16* __restrict__ xattn, int t) {
  int b = blockIdx.y;
  int e = blockIdx.x * 256 + threadIdx.x;  // gridDim.x = 2
  const float* aw  = aligns + ((size_t)b * CTD + t) * CTE;
  const float* mem = memory + (size_t)b * CTE * CE + e;
  float acc = 0.f;
  for (int tc = 0; tc < CTE; ++tc) acc += aw[tc] * mem[(size_t)tc * CE];
  ctx[(size_t)b * CE + e] = acc;
  xattn[(size_t)b * CKA + CP + e] = (_Float16)acc;
}

// gates = X@Wih^T + H@Whh^T + b_ih + b_hh  (WMMA).
// X[32,KX] and H[32,CD] staged ONCE per workgroup into LDS via async global->LDS b128
// copies (ASYNCcnt), +8-half row pitch to spread ds_load_b128 across LDS banks;
// all 8 waves then share the staged A operands.
template <int KX>
__global__ void k_lstm_gates(const _Float16* __restrict__ X, const _Float16* __restrict__ Wxt,
                             const _Float16* __restrict__ H, const _Float16* __restrict__ Wht,
                             const float* __restrict__ bih, const float* __restrict__ bhh,
                             float* __restrict__ gates) {
  constexpr int PX = KX + 8;   // padded pitch (halves)
  constexpr int PH = CD + 8;
  __shared__ __align__(16) _Float16 sX[32 * PX];
  __shared__ __align__(16) _Float16 sH[32 * PH];
  int tid  = threadIdx.x;
  int lane = tid & 31;
  int wave = tid >> 5;
  int ntile = blockIdx.x * 8 + wave;  // 0..255

  // async stage X and H into LDS (16B granules, uniform trip counts)
  constexpr int CHX = 32 * (KX / 8);
  for (int c = tid; c < CHX; c += 256) {
    int row = c / (KX / 8), col = (c % (KX / 8)) * 8;
    async_g2l_b128((uint32_t)(uintptr_t)&sX[row * PX + col], X + (size_t)row * KX + col);
  }
  constexpr int CHH = 32 * (CD / 8);
  for (int c = tid; c < CHH; c += 256) {
    int row = c / (CD / 8), col = (c % (CD / 8)) * 8;
    async_g2l_b128((uint32_t)(uintptr_t)&sH[row * PH + col], H + (size_t)row * CD + col);
  }
  wait_asynccnt0();
  __syncthreads();

  v8f c0 = {}, c1 = {};
  constexpr int KTx = KX / 32;
  for (int kt = 0; kt < KTx; ++kt) {
    v16h bfr = load_b_tile(Wxt, ntile * KTx + kt, lane);
    v16h a0  = load_a_f16(sX, PX, 0, kt, lane);
    v16h a1  = load_a_f16(sX, PX, 16, kt, lane);
    c0 = __builtin_amdgcn_wmma_f32_16x16x32_f16(false, a0, false, bfr, (short)0, c0, false, false);
    c1 = __builtin_amdgcn_wmma_f32_16x16x32_f16(false, a1, false, bfr, (short)0, c1, false, false);
  }
  constexpr int KTh = CD / 32;  // 32
  for (int kt = 0; kt < KTh; ++kt) {
    v16h bfr = load_b_tile(Wht, ntile * KTh + kt, lane);
    v16h a0  = load_a_f16(sH, PH, 0, kt, lane);
    v16h a1  = load_a_f16(sH, PH, 16, kt, lane);
    c0 = __builtin_amdgcn_wmma_f32_16x16x32_f16(false, a0, false, bfr, (short)0, c0, false, false);
    c1 = __builtin_amdgcn_wmma_f32_16x16x32_f16(false, a1, false, bfr, (short)0, c1, false, false);
  }
  int n = ntile * 16 + (lane & 15);
  float bias = bih[n] + bhh[n];
  int mlo = (lane & 16) ? 8 : 0;
#pragma unroll
  for (int r = 0; r < 8; ++r) {
    gates[(size_t)(mlo + r) * CG + n]      = c0[r] + bias;
    gates[(size_t)(16 + mlo + r) * CG + n] = c1[r] + bias;
  }
}

__global__ void k_lstm_cell(const float* __restrict__ gates, float* __restrict__ h,
                            float* __restrict__ c, _Float16* __restrict__ h16) {
  int b = blockIdx.x;
  int d = blockIdx.y * 256 + threadIdx.x;  // gridDim.y = 4
  const float* g = gates + (size_t)b * CG;
  float gi = g[d], gf = g[CD + d], gg = g[2 * CD + d], go = g[3 * CD + d];
  size_t idx = (size_t)b * CD + d;
  float cc = sigmoidf_(gf) * c[idx] + sigmoidf_(gi) * tanhf(gg);
  float hh = sigmoidf_(go) * tanhf(cc);
  c[idx] = cc; h[idx] = hh; h16[idx] = (_Float16)hh;
}

__global__ void k_project(const float* __restrict__ dh, const float* __restrict__ ctx,
                          const float* __restrict__ proj_w, const float* __restrict__ proj_b,
                          const float* __restrict__ gate_w, const float* __restrict__ gate_b,
                          float* __restrict__ out, int t) {
  int b = blockIdx.x, tid = threadIdx.x;  // 128 threads
  const float* dhr = dh + (size_t)b * CD;
  const float* cxr = ctx + (size_t)b * CE;
  if (tid < CM) {
    const float* w = proj_w + (size_t)tid * (CD + CE);
    float s = proj_b[tid];
    for (int k = 0; k < CD; ++k) s += w[k] * dhr[k];
    for (int k = 0; k < CE; ++k) s += w[CD + k] * cxr[k];
    out[((size_t)b * CTD + t) * CM + tid] = s;          // mels [B,TDEC,M]
  } else if (tid == CM) {
    float s = gate_b[0];
    for (int k = 0; k < CD; ++k) s += gate_w[k] * dhr[k];
    for (int k = 0; k < CE; ++k) s += gate_w[CD + k] * cxr[k];
    out[(size_t)CB * CTD * CM + (size_t)b * CTD + t] = s;  // gates [B,TDEC]
  }
}

// ================= host =================
extern "C" void kernel_launch(void* const* d_in, const int* in_sizes, int n_in,
                              void* d_out, int out_size, void* d_ws, size_t ws_size,
                              hipStream_t stream) {
  (void)in_sizes; (void)n_in; (void)out_size; (void)ws_size;
  const float* memory      = (const float*)d_in[0];
  const float* dec_inputs  = (const float*)d_in[1];
  const unsigned char* msk = (const unsigned char*)d_in[2];
  const float* prenet_w1   = (const float*)d_in[3];
  const float* prenet_w2   = (const float*)d_in[4];
  const float* q_w         = (const float*)d_in[5];
  const float* mem_w       = (const float*)d_in[6];
  const float* loc_conv_w  = (const float*)d_in[7];
  const float* loc_w       = (const float*)d_in[8];
  const float* v_w         = (const float*)d_in[9];
  const float* v_b         = (const float*)d_in[10];
  const float* attn_w_ih   = (const float*)d_in[11];
  const float* attn_w_hh   = (const float*)d_in[12];
  const float* attn_b_ih   = (const float*)d_in[13];
  const float* attn_b_hh   = (const float*)d_in[14];
  const float* dec_w_ih    = (const float*)d_in[15];
  const float* dec_w_hh    = (const float*)d_in[16];
  const float* dec_b_ih    = (const float*)d_in[17];
  const float* dec_b_hh    = (const float*)d_in[18];
  const float* proj_w      = (const float*)d_in[19];
  const float* proj_b      = (const float*)d_in[20];
  const float* gate_w      = (const float*)d_in[21];
  const float* gate_b      = (const float*)d_in[22];

  uint8_t* ws = (uint8_t*)d_ws;
  size_t off = 0;
  auto alloc = [&](size_t bytes) -> void* {
    void* r = ws + off;
    off += (bytes + 255) & ~(size_t)255;
    return r;
  };
  _Float16* WXA  = (_Float16*)alloc((size_t)CG * CKA * 2);
  _Float16* WHA  = (_Float16*)alloc((size_t)CG * CD * 2);
  _Float16* WXD  = (_Float16*)alloc((size_t)CG * CD * 2);
  _Float16* WHD  = (_Float16*)alloc((size_t)CG * CD * 2);
  _Float16* WMEM = (_Float16*)alloc((size_t)CA * CE * 2);
  float* U     = (float*)alloc((size_t)CA * CLK * 4);
  float* PM    = (float*)alloc((size_t)CB * CTE * CA * 4);
  float* AH    = (float*)alloc((size_t)CB * CD * 4);
  float* AC    = (float*)alloc((size_t)CB * CD * 4);
  float* DH    = (float*)alloc((size_t)CB * CD * 4);
  float* DC    = (float*)alloc((size_t)CB * CD * 4);
  _Float16* AH16  = (_Float16*)alloc((size_t)CB * CD * 2);
  _Float16* DH16  = (_Float16*)alloc((size_t)CB * CD * 2);
  _Float16* XATTN = (_Float16*)alloc((size_t)CB * CKA * 2);
  float* CUMW  = (float*)alloc((size_t)CB * CTE * 4);
  float* CTX   = (float*)alloc((size_t)CB * CE * 4);
  float* GATES = (float*)alloc((size_t)CB * CG * 4);

  auto swz = [&](const float* W, _Float16* dst, int N, int K) {
    int total = (N / 16) * (K / 32) * 32;
    k_swizzle<<<(total + 255) / 256, 256, 0, stream>>>(W, dst, N, K);
  };
  // ---- one-time prep (runs every call; ws is poisoned by harness) ----
  swz(attn_w_ih, WXA, CG, CKA);
  swz(attn_w_hh, WHA, CG, CD);
  swz(dec_w_ih,  WXD, CG, CD);
  swz(dec_w_hh,  WHD, CG, CD);
  swz(mem_w,     WMEM, CA, CE);
  k_uloc<<<1, 128, 0, stream>>>(loc_w, loc_conv_w, U);
  k_pm_wmma<<<(CB * CTE) / 32, 256, 0, stream>>>(memory, WMEM, PM);
  k_init<<<128, 256, 0, stream>>>(AH, AC, DH, DC, AH16, DH16, CUMW);

  float* out = (float*)d_out;
  float* aligns = out + (size_t)CB * CTD * CM + (size_t)CB * CTD;

  // ---- 128 sequential decode steps; dependency via stream order ----
  for (int t = 0; t < CTD; ++t) {
    k_prenet<<<CB, CP, 0, stream>>>(dec_inputs, prenet_w1, prenet_w2, XATTN, t);
    k_attention<<<CB, 256, 0, stream>>>(AH, q_w, PM, U, v_w, v_b, msk, CUMW, aligns, t);
    dim3 gctx(CE / 256, CB);
    k_context<<<gctx, 256, 0, stream>>>(aligns, memory, CTX, XATTN, t);
    k_lstm_gates<CKA><<<32, 256, 0, stream>>>(XATTN, WXA, AH16, WHA, attn_b_ih, attn_b_hh, GATES);
    dim3 gcell(CB, CD / 256);
    k_lstm_cell<<<gcell, 256, 0, stream>>>(GATES, AH, AC, AH16);
    k_lstm_gates<CD><<<32, 256, 0, stream>>>(AH16, WXD, DH16, WHD, dec_b_ih, dec_b_hh, GATES);
    k_lstm_cell<<<gcell, 256, 0, stream>>>(GATES, DH, DC, DH16);
    k_project<<<CB, 128, 0, stream>>>(DH, CTX, proj_w, proj_b, gate_w, gate_b, out, t);
  }
}
